// SS2D_59416577573148
// MI455X (gfx1250) — compile-verified, matrix operand order checked by reading, not run on hardware
//
#include <hip/hip_runtime.h>
#include <hip/hip_bf16.h>

// ---------------------------------------------------------------------------
// SS2D forward for MI455X (gfx1250, wave32, WMMA).
// Constants match the reference module.
// ---------------------------------------------------------------------------
#define B_SZ   2
#define H_DIM  32
#define W_DIM  32
#define L_LEN  1024          // H*W
#define DIM_   128
#define E_     256           // 2*DIM
#define MID_   16            // E/16
#define NSTATE 16            // DSTATE
#define RANK_  8
#define NDIR_  4
#define ROWS_  (B_SZ * L_LEN)   // 2048

typedef __attribute__((ext_vector_type(16))) _Float16 v16h;
typedef __attribute__((ext_vector_type(8)))  float    v8f;
typedef __attribute__((ext_vector_type(4)))  float    v4f;

// ---------------------------------------------------------------------------
// sass(): generate the 4 scan orders + direction codes (deterministic, tiny).
// ---------------------------------------------------------------------------
__global__ void gen_sass_kernel(int* __restrict__ ord, int* __restrict__ dirs) {
  if (threadIdx.x != 0 || blockIdx.x != 0) return;
  const int H = H_DIM, W = W_DIM, L = L_LEN;

  // direction 0: row snake from bottom
  {
    int i, j, jd;                       // jd: 1=right, 0=left
    if (H & 1) { i = H - 1; j = W - 1; jd = 0; }
    else       { i = H - 1; j = 0;     jd = 1; }
    int cnt = 0; dirs[0 * L] = 0;
    while (i > -1) {
      ord[0 * L + cnt] = i * W + j;
      int dc;
      if (jd == 1) { if (j < W - 1) { j++; dc = 1; } else { i--; dc = 3; jd = 0; } }
      else         { if (j > 0)     { j--; dc = 2; } else { i--; dc = 3; jd = 1; } }
      if (cnt + 1 < L) dirs[0 * L + cnt + 1] = dc;
      cnt++;
    }
  }
  // direction 1: column snake from left
  {
    int i = 0, j = 0, idd = 1;          // idd: 1=down, 0=up
    int cnt = 0; dirs[1 * L] = 0;
    while (j < W) {
      ord[1 * L + cnt] = i * W + j;
      int dc;
      if (idd == 1) { if (i < H - 1) { i++; dc = 4; } else { j++; dc = 1; idd = 0; } }
      else          { if (i > 0)     { i--; dc = 3; } else { j++; dc = 1; idd = 1; } }
      if (cnt + 1 < L) dirs[1 * L + cnt + 1] = dc;
      cnt++;
    }
  }
  // directions 2 and 3: zig-zag diagonals (dir 3 mirrors columns)
  for (int m = 0; m < 2; ++m) {
    const int K = 2 + m;
    int cnt = 0; dirs[K * L] = 0;
    for (int diag = 0; diag < H + W - 1; ++diag) {
      if ((diag & 1) == 0) {
        int lim = (diag + 1 < H) ? diag + 1 : H;
        for (int ii = 0; ii < lim; ++ii) {
          int i = ii, j = diag - ii;
          if (j < W) {
            int jj2 = (m == 0) ? j : (W - j - 1);
            ord[K * L + cnt] = i * W + jj2;
            int dc = (j == diag) ? 1 : 4;
            if (cnt + 1 < L) dirs[K * L + cnt + 1] = dc;
            cnt++;
          }
        }
      } else {
        int lim = (diag + 1 < W) ? diag + 1 : W;
        for (int jj = 0; jj < lim; ++jj) {
          int j = jj, i = diag - jj;
          if (i < H) {
            int jj2 = (m == 0) ? j : (W - j - 1);
            ord[K * L + cnt] = i * W + jj2;
            int dc = (i == diag) ? 4 : 1;
            if (cnt + 1 < L) dirs[K * L + cnt + 1] = dc;
            cnt++;
          }
        }
      }
    }
  }
}

// ---------------------------------------------------------------------------
// Generic wave-per-16x16-tile WMMA GEMM, fully compile-time shapes/strides.
//   C = A(MxK, f32, row stride LDA) * B(KxN, f32, elem (k,n) at k*BRS + n*BCS)
// K must be a multiple of 32, or exactly 16 (zero-padded to one WMMA; the
// valid/zero split is lane-independent, so no divergence — EXEC stays ~0).
//
// 16-bit A 16x32 fragment (ISA 7.12.2): lane = half_sel*16 + m,
//   element e -> K = (e<8 ? e : e+8) + 8*half_sel
// i.e. two contiguous 8-float runs per lane -> four b128 loads + cvt.
//
// EPI 0: split cols [0,256)->C0 (xm), [256,512)->C1 (z)
// EPI 1: +bias, store NCHW h_mid[(b*MID+col)*L + hw]
// EPI 2: SiLU then store row-major
// EPI 3: plain store row-major
// ---------------------------------------------------------------------------
template <int EPI, int LDA, int BRS, int BCS, int N, int K>
__global__ void wmma_gemm_kernel(const float* __restrict__ A,
                                 const float* __restrict__ B,
                                 const float* __restrict__ bias,
                                 float* __restrict__ C0,
                                 float* __restrict__ C1) {
  static_assert((K % 32 == 0) || (K == 16), "K must be 32n or 16");
  const int wave = blockIdx.x * (blockDim.x >> 5) + (threadIdx.x >> 5);
  const int lane = threadIdx.x & 31;
  constexpr int NT = N >> 4;
  const int tm = wave / NT;
  const int tn = wave % NT;
  const int row0 = tm << 4, col0 = tn << 4;
  const int half_sel = lane >> 4;
  const int mn = lane & 15;

  v8f acc = {};
#pragma unroll
  for (int k0 = 0; k0 < K; k0 += 32) {
    // ---- A fragment: two contiguous 8-float runs, 32B-aligned ----
    const float* arow = A + (size_t)(row0 + mn) * LDA + k0 + half_sel * 8;
    v4f a0 = *(const v4f*)(arow);
    v4f a1 = *(const v4f*)(arow + 4);
    v4f a2 = {0.f, 0.f, 0.f, 0.f};
    v4f a3 = {0.f, 0.f, 0.f, 0.f};
    if (K != 16) { a2 = *(const v4f*)(arow + 16); a3 = *(const v4f*)(arow + 20); }
    v16h af;
#pragma unroll
    for (int e = 0; e < 4; ++e) {
      af[e]      = (_Float16)a0[e];
      af[4 + e]  = (_Float16)a1[e];
      af[8 + e]  = (_Float16)a2[e];
      af[12 + e] = (_Float16)a3[e];
    }
    // ---- B fragment ----
    v16h bf;
    if (BRS == 1) {              // K-contiguous weights ([N,K] storage)
      const float* bcol = B + (size_t)(col0 + mn) * BCS + k0 + half_sel * 8;
      v4f b0 = *(const v4f*)(bcol);
      v4f b1 = *(const v4f*)(bcol + 4);
      v4f b2 = {0.f, 0.f, 0.f, 0.f};
      v4f b3 = {0.f, 0.f, 0.f, 0.f};
      if (K != 16) { b2 = *(const v4f*)(bcol + 16); b3 = *(const v4f*)(bcol + 20); }
#pragma unroll
      for (int e = 0; e < 4; ++e) {
        bf[e]      = (_Float16)b0[e];
        bf[4 + e]  = (_Float16)b1[e];
        bf[8 + e]  = (_Float16)b2[e];
        bf[12 + e] = (_Float16)b3[e];
      }
    } else {                     // row-major [K,N]: columns coalesce over lanes
      const float* bptr = B + (size_t)(k0 + half_sel * 8) * BRS
                            + (size_t)(col0 + mn) * BCS;
#pragma unroll
      for (int e = 0; e < 8; ++e) bf[e] = (_Float16)bptr[e * BRS];
      if (K != 16) {
        const float* bptr2 = bptr + 16 * BRS;
#pragma unroll
        for (int e = 0; e < 8; ++e) bf[8 + e] = (_Float16)bptr2[e * BRS];
      } else {
#pragma unroll
        for (int e = 0; e < 8; ++e) bf[8 + e] = (_Float16)0.f;
      }
    }
    acc = __builtin_amdgcn_wmma_f32_16x16x32_f16(false, af, false, bf,
                                                 (short)0, acc, false, false);
  }

  const int col = col0 + mn;
#pragma unroll
  for (int r = 0; r < 8; ++r) {
    const int row = row0 + r + half_sel * 8;   // f32 C/D layout (ISA 7.12.2)
    float v = acc[r];
    if (EPI == 0) {
      if (col < E_) C0[(size_t)row * E_ + col] = v;
      else          C1[(size_t)row * E_ + (col - E_)] = v;
    } else if (EPI == 1) {
      int bb = row >> 10, hw = row & (L_LEN - 1);
      C0[((size_t)bb * MID_ + col) * L_LEN + hw] = v + bias[col];
    } else if (EPI == 2) {
      C0[(size_t)row * N + col] = v / (1.0f + __expf(-v));
    } else {
      C0[(size_t)row * N + col] = v;
    }
  }
}

// ---------------------------------------------------------------------------
// 3x3 depthwise conv (pad 1) on the 16-channel mid feature map.
// hmid: [B][MID][H][W] -> dwout: [B*L][MID] row-major for the next GEMM.
// ---------------------------------------------------------------------------
__global__ void dw_kernel(const float* __restrict__ hmid,
                          const float* __restrict__ w,
                          float* __restrict__ dwout) {
  int gid = blockIdx.x * blockDim.x + threadIdx.x;   // ROWS_*MID_
  int ch = gid & (MID_ - 1);
  int r  = gid >> 4;
  int b  = r >> 10;
  int hw = r & (L_LEN - 1);
  int yy = hw >> 5, xx = hw & 31;
  const float* plane = hmid + ((size_t)b * MID_ + ch) * L_LEN;
  const float* wk = w + ch * 9;
  float s = 0.0f;
#pragma unroll
  for (int dy = 0; dy < 3; ++dy) {
    int y2 = yy + dy - 1;
    if (y2 < 0 || y2 >= H_DIM) continue;
#pragma unroll
    for (int dx = 0; dx < 3; ++dx) {
      int x2 = xx + dx - 1;
      if (x2 < 0 || x2 >= W_DIM) continue;
      s += plane[y2 * W_DIM + x2] * wk[dy * 3 + dx];
    }
  }
  dwout[(size_t)r * MID_ + ch] = s;
}

// x_dbl = x_conv @ x_proj_w  ([2048,256] x [256,40])
__global__ void xdbl_kernel(const float* __restrict__ xconv,
                            const float* __restrict__ W,
                            float* __restrict__ xdbl) {
  int gid = blockIdx.x * blockDim.x + threadIdx.x;   // ROWS_*40
  int c = gid % 40;
  int r = gid / 40;
  const float* row = xconv + (size_t)r * E_;
  float s = 0.0f;
  for (int k = 0; k < E_; ++k) s += row[k] * W[k * 40 + c];
  xdbl[(size_t)r * 40 + c] = s;
}

// dt = dt_lr @ dt_proj_w + dt_proj_b  ([2048,8] x [8,256])
__global__ void dtproj_kernel(const float* __restrict__ xdbl,
                              const float* __restrict__ W,
                              const float* __restrict__ bias,
                              float* __restrict__ dt) {
  int r = blockIdx.x;
  int e = threadIdx.x;
  const float* dl = xdbl + (size_t)r * 40;
  float s = bias[e];
#pragma unroll
  for (int k = 0; k < RANK_; ++k) s += dl[k] * W[k * E_ + e];
  dt[(size_t)r * E_ + e] = s;
}

// ---------------------------------------------------------------------------
// Selective scan: one lane per (b,dir,e) channel, N=16 state in VGPRs.
// Block = one (b,dir): 256 threads = 8 wave32. B/C (16 each) staged in LDS.
// ---------------------------------------------------------------------------
__global__ void scan_kernel(const float* __restrict__ xconv,
                            const float* __restrict__ dt,
                            const float* __restrict__ xdbl,
                            const float* __restrict__ A_log,
                            const float* __restrict__ dirB,
                            const float* __restrict__ dtb,
                            const float* __restrict__ Dp,
                            const int* __restrict__ ord,
                            const int* __restrict__ dirs,
                            float* __restrict__ ydir) {
  const int b = blockIdx.x >> 2;
  const int k = blockIdx.x & 3;
  const int e = threadIdx.x;

  __shared__ float sB[NSTATE];
  __shared__ float sC[NSTATE];

  float h[NSTATE];
  float Arow[NSTATE];
#pragma unroll
  for (int n = 0; n < NSTATE; ++n) {
    h[n] = 0.0f;
    Arow[n] = -__expf(A_log[e * NSTATE + n]);   // A = -exp(A_log)
  }
  const float dt_bias = dtb[e];
  const float Dv = Dp[e];
  const int* O  = ord  + k * L_LEN;
  const int* DD = dirs + k * L_LEN;

  for (int l = 0; l < L_LEN; ++l) {
    const int idx = O[l];
    if (threadIdx.x < 32) {
      int n = threadIdx.x & 15;
      const float* xd = xdbl + ((size_t)b * L_LEN + idx) * 40;
      if (threadIdx.x < 16) sB[n] = xd[RANK_ + n] + dirB[DD[l] * NSTATE + n];
      else                  sC[n] = xd[RANK_ + NSTATE + n];
    }
    if (l + 1 < L_LEN) {                    // hide the indirect-gather latency
      int nidx = O[l + 1];
      __builtin_prefetch(dt    + ((size_t)b * L_LEN + nidx) * E_ + e, 0, 0);
      __builtin_prefetch(xconv + ((size_t)b * L_LEN + nidx) * E_ + e, 0, 0);
    }
    __syncthreads();

    float dtv = dt[((size_t)b * L_LEN + idx) * E_ + e] + dt_bias;
    float delta = (dtv > 20.0f) ? dtv : __logf(1.0f + __expf(dtv));  // softplus
    float u  = xconv[((size_t)b * L_LEN + idx) * E_ + e];
    float du = delta * u;
    float y = 0.0f;
#pragma unroll
    for (int n = 0; n < NSTATE; ++n) {
      float dA = __expf(delta * Arow[n]);
      h[n] = dA * h[n] + du * sB[n];
      y += h[n] * sC[n];
    }
    ydir[(((size_t)k * B_SZ + b) * L_LEN + l) * E_ + e] = y + Dv * u;
    __syncthreads();
  }
}

// y_sum[b,j,e] = sum_k y_scan[b,k,O[k,j],e]; then gate with SiLU(z).
__global__ void fuse_kernel(const float* __restrict__ ydir,
                            const int* __restrict__ ord,
                            const float* __restrict__ z,
                            float* __restrict__ g) {
  int r = blockIdx.x;            // b*L + j
  int e = threadIdx.x;
  int b = r >> 10;
  int j = r & (L_LEN - 1);
  float s = 0.0f;
#pragma unroll
  for (int k = 0; k < NDIR_; ++k) {
    int pos = ord[k * L_LEN + j];
    s += ydir[(((size_t)k * B_SZ + b) * L_LEN + pos) * E_ + e];
  }
  float zv = z[(size_t)r * E_ + e];
  g[(size_t)r * E_ + e] = s * (zv / (1.0f + __expf(-zv)));
}

// ---------------------------------------------------------------------------
extern "C" void kernel_launch(void* const* d_in, const int* in_sizes, int n_in,
                              void* d_out, int out_size, void* d_ws, size_t ws_size,
                              hipStream_t stream) {
  const float* x      = (const float*)d_in[0];   // [2,1024,128]
  const float* w_in   = (const float*)d_in[1];   // [128,512]
  const float* pw1w   = (const float*)d_in[2];   // [16,256,1,1]
  const float* pw1b   = (const float*)d_in[3];   // [16]
  const float* dww    = (const float*)d_in[4];   // [16,1,3,3]
  const float* pw2w   = (const float*)d_in[5];   // [256,16,1,1]
  const float* xprojw = (const float*)d_in[6];   // [256,40]
  const float* dtw    = (const float*)d_in[7];   // [8,256]
  const float* dtb    = (const float*)d_in[8];   // [256]
  const float* A_log  = (const float*)d_in[9];   // [256,16]
  const float* Dp     = (const float*)d_in[10];  // [256]
  const float* wout   = (const float*)d_in[11];  // [256,128]
  const float* dirB   = (const float*)d_in[12];  // [5,16]
  float* out = (float*)d_out;                    // [2,1024,128]

  char* p = (char*)d_ws;
  auto carve = [&](size_t bytes) -> char* {
    char* r = p;
    p += (bytes + 255) & ~(size_t)255;
    return r;
  };
  int*   ord   = (int*)  carve((size_t)NDIR_ * L_LEN * 4);
  int*   dirs  = (int*)  carve((size_t)NDIR_ * L_LEN * 4);
  float* xm    = (float*)carve((size_t)ROWS_ * E_ * 4);
  float* zb    = (float*)carve((size_t)ROWS_ * E_ * 4);
  float* hmid  = (float*)carve((size_t)B_SZ * MID_ * L_LEN * 4);
  float* dwout = (float*)carve((size_t)ROWS_ * MID_ * 4);
  float* xconv = (float*)carve((size_t)ROWS_ * E_ * 4);
  float* xdbl  = (float*)carve((size_t)ROWS_ * 40 * 4);
  float* dtbuf = (float*)carve((size_t)ROWS_ * E_ * 4);
  float* ydir  = (float*)carve((size_t)NDIR_ * ROWS_ * E_ * 4);
  float* g     = (float*)carve((size_t)ROWS_ * E_ * 4);
  (void)in_sizes; (void)n_in; (void)out_size; (void)ws_size;

  // 1) scan orders / direction codes
  gen_sass_kernel<<<1, 1, 0, stream>>>(ord, dirs);

  // 2) in_proj: [2048,128]x[128,512] -> xm | z    (4096 tiles, 4 waves/block)
  wmma_gemm_kernel<0, DIM_, 2 * E_, 1, 2 * E_, DIM_>
      <<<1024, 128, 0, stream>>>(x, w_in, nullptr, xm, zb);

  // 3) pw1 1x1: [2048,256]x[256,16] (+bias) -> hmid NCHW   (128 tiles)
  wmma_gemm_kernel<1, E_, 1, E_, MID_, E_>
      <<<32, 128, 0, stream>>>(xm, pw1w, pw1b, hmid, nullptr);

  // 4) 3x3 depthwise
  dw_kernel<<<(ROWS_ * MID_) / 256, 256, 0, stream>>>(hmid, dww, dwout);

  // 5) pw2 1x1 + SiLU: [2048,16]x[16,256] -> x_conv  (K=16 zero-padded to 32)
  wmma_gemm_kernel<2, MID_, 1, MID_, E_, MID_>
      <<<512, 128, 0, stream>>>(dwout, pw2w, nullptr, xconv, nullptr);

  // 6) x_dbl projection [2048,256]x[256,40]
  xdbl_kernel<<<(ROWS_ * 40) / 256, 256, 0, stream>>>(xconv, xprojw, xdbl);

  // 7) dt projection [2048,8]x[8,256] + bias
  dtproj_kernel<<<ROWS_, E_, 0, stream>>>(xdbl, dtw, dtb, dtbuf);

  // 8) 4-direction selective scan (8 blocks x 256 threads = 64 waves)
  scan_kernel<<<B_SZ * NDIR_, E_, 0, stream>>>(xconv, dtbuf, xdbl, A_log, dirB,
                                               dtb, Dp, ord, dirs, ydir);

  // 9) direction gather + SiLU(z) gate
  fuse_kernel<<<ROWS_, E_, 0, stream>>>(ydir, ord, zb, g);

  // 10) out_proj: [2048,256]x[256,128] -> out   (1024 tiles)
  wmma_gemm_kernel<3, E_, DIM_, 1, DIM_, E_>
      <<<256, 128, 0, stream>>>(g, wout, nullptr, out, nullptr);
}
